// AttentionBlock_39831526703747
// MI455X (gfx1250) — matrix operand order repaired and checked by reading, not compiled, not run on hardware
//
#include <hip/hip_runtime.h>

#define BATCH 8
#define CH 256
#define HW 4096
#define CPG 32
#define GROUP_ELEMS (CPG * HW)
#define GN_EPS 1e-5f
#define ATT_SCALE 0.0625f  // 1/sqrt(256)

typedef __attribute__((ext_vector_type(16))) __bf16 bf16x16;
typedef __attribute__((ext_vector_type(8)))  __bf16 bf16x8;
typedef __attribute__((ext_vector_type(8)))  float  f32x8;
typedef __attribute__((ext_vector_type(4)))  float  f32x4;

static __device__ __forceinline__ int lane_id() { return (int)(threadIdx.x & 31u); }

static __device__ __forceinline__ f32x8 wmma_bf16(bf16x16 a, bf16x16 b, f32x8 c) {
  return __builtin_amdgcn_wmma_f32_16x16x32_bf16(false, a, false, b, (short)0, c, false, false);
}

static __device__ __forceinline__ void zero8(f32x8& v) {
#pragma unroll
  for (int i = 0; i < 8; i++) v[i] = 0.0f;
}

// ---- A-fragment (16x32, M x K): lane = M row; K = {coff..coff+7, coff+16..coff+23}
// p = row start (for this lane's M) + k0, contiguous in K.
static __device__ __forceinline__ bf16x16 fragA_b16(const __bf16* p) {
  int coff = (lane_id() < 16) ? 0 : 8;
  bf16x8 h0 = *(const bf16x8*)(p + coff);
  bf16x8 h1 = *(const bf16x8*)(p + coff + 16);
  bf16x16 f;
#pragma unroll
  for (int i = 0; i < 8; i++) { f[i] = h0[i]; f[i + 8] = h1[i]; }
  return f;
}

static __device__ __forceinline__ bf16x16 fragA_f32(const float* p) {
  int coff = (lane_id() < 16) ? 0 : 8;
  f32x4 a0 = *(const f32x4*)(p + coff);
  f32x4 a1 = *(const f32x4*)(p + coff + 4);
  f32x4 b0 = *(const f32x4*)(p + coff + 16);
  f32x4 b1 = *(const f32x4*)(p + coff + 20);
  bf16x16 f;
#pragma unroll
  for (int i = 0; i < 4; i++) {
    f[i] = (__bf16)a0[i]; f[4 + i] = (__bf16)a1[i];
    f[8 + i] = (__bf16)b0[i]; f[12 + i] = (__bf16)b1[i];
  }
  return f;
}

// ---- B-fragment (32x16, K x N): lane = N col; K = {koff..koff+7, koff+16..koff+23},
// koff = lane<16 ? 0 : 8.  p = row start (for this lane's N) + k0.
static __device__ __forceinline__ bf16x16 fragB_b16(const __bf16* p) {
  int koff = (lane_id() < 16) ? 0 : 8;
  bf16x8 h0 = *(const bf16x8*)(p + koff);
  bf16x8 h1 = *(const bf16x8*)(p + koff + 16);
  bf16x16 f;
#pragma unroll
  for (int i = 0; i < 8; i++) { f[i] = h0[i]; f[i + 8] = h1[i]; }
  return f;
}

static __device__ __forceinline__ bf16x16 fragB_f32(const float* p) {
  int koff = (lane_id() < 16) ? 0 : 8;
  bf16x16 f;
#pragma unroll
  for (int i = 0; i < 4; i++) {
    f32x4 a = *(const f32x4*)(p + koff + 4 * (i & 1) + 16 * (i >> 1));
#pragma unroll
    for (int j = 0; j < 4; j++) f[4 * i + j] = (__bf16)a[j];
  }
  return f;
}

// ---- B-fragment gathered from a [K][col] f32 layout (column access is a
// per-K broadcast row read: lanes 0-15 hit 64B-contiguous floats).
// base points at (k0, colbase); this lane's col = lane&15.
static __device__ __forceinline__ bf16x16 fragB_f32_gather(const float* base, long kstride) {
  int lane = lane_id();
  int koff = (lane < 16) ? 0 : 8;
  const float* p = base + (lane & 15);
  bf16x16 f;
#pragma unroll
  for (int e = 0; e < 16; e++) f[e] = (__bf16)p[(long)(koff + (e & 7) + 16 * (e >> 3)) * kstride];
  return f;
}

// ---- A-fragment gathered from a [K][col] f32 layout.
static __device__ __forceinline__ bf16x16 fragA_f32_gather(const float* base, long kstride) {
  int lane = lane_id();
  int coff = (lane < 16) ? 0 : 8;
  const float* p = base + (lane & 15);
  bf16x16 f;
#pragma unroll
  for (int e = 0; e < 8; e++) f[e] = (__bf16)p[(long)(coff + e) * kstride];
#pragma unroll
  for (int e = 0; e < 8; e++) f[8 + e] = (__bf16)p[(long)(coff + 16 + e) * kstride];
  return f;
}

// ================= Kernel 1: GroupNorm, writes xn fp32 in [b][c][t] =========
__global__ __launch_bounds__(256) void gn_kernel(const float* __restrict__ x,
                                                 const float* __restrict__ gamma,
                                                 const float* __restrict__ beta,
                                                 float* __restrict__ xn) {
  int bg = blockIdx.x;
  int b = bg >> 3, g = bg & 7;
  const float* xb = x + ((long)b * CH + g * CPG) * HW;
  float s = 0.0f, ss = 0.0f;
  for (int i = threadIdx.x; i < GROUP_ELEMS; i += 256) {
    float v = xb[i];
    s += v; ss += v * v;
  }
  __shared__ float red[512];
  red[threadIdx.x] = s;
  red[256 + threadIdx.x] = ss;
  __syncthreads();
  for (int off = 128; off > 0; off >>= 1) {
    if ((int)threadIdx.x < off) {
      red[threadIdx.x] += red[threadIdx.x + off];
      red[256 + threadIdx.x] += red[256 + threadIdx.x + off];
    }
    __syncthreads();
  }
  float mean = red[0] * (1.0f / GROUP_ELEMS);
  float var = red[256] * (1.0f / GROUP_ELEMS) - mean * mean;
  float rstd = rsqrtf(var + GN_EPS);
  float* xnb = xn + ((long)b * CH + g * CPG) * HW;
  for (int i = threadIdx.x; i < GROUP_ELEMS; i += 256) {
    int c = g * CPG + (i >> 12);
    xnb[i] = (xb[i] - mean) * rstd * gamma[c] + beta[c];
  }
}

// ====== Kernel 2a: q,k GEMM. D[t,o] = sum_c xn[c,t] W[o,c] + b[o]  ==========
// Orientation: M=t, N=o.  q,k stored bf16 [b][t][c].
__global__ __launch_bounds__(256) void qk_gemm(const float* __restrict__ xn,
                                               const float* __restrict__ qkv_w,
                                               const float* __restrict__ qkv_b,
                                               __bf16* __restrict__ q,
                                               __bf16* __restrict__ k) {
  int wid = (int)((blockIdx.x * blockDim.x + threadIdx.x) >> 5);
  int b = wid >> 11;            // 2048 waves per batch
  int rem = wid & 2047;
  int tbase = (rem & 255) * 16; // 256 t-tiles of 16
  int obase = (rem >> 8) * 64;  // 8 o-tiles of 64 (over 512 q+k channels)
  int lane = lane_id();
  f32x8 acc[4];
#pragma unroll
  for (int n = 0; n < 4; n++) zero8(acc[n]);
  const float* xnb = xn + (long)b * CH * HW;
#pragma unroll
  for (int ks = 0; ks < 8; ks++) {
    int c0 = ks * 32;
    bf16x16 af = fragA_f32_gather(xnb + (long)c0 * HW + tbase, HW);
#pragma unroll
    for (int n = 0; n < 4; n++) {
      bf16x16 bf = fragB_f32(qkv_w + (long)(obase + n * 16 + (lane & 15)) * CH + c0);
      acc[n] = wmma_bf16(af, bf, acc[n]);
    }
  }
  int rbase = (lane < 16) ? 0 : 8;
#pragma unroll
  for (int n = 0; n < 4; n++) {
    int o = obase + n * 16 + (lane & 15);
    float bias = qkv_b[o];
    __bf16* dst = (o < CH) ? q : k;
    int oc = (o < CH) ? o : (o - CH);
#pragma unroll
    for (int r = 0; r < 8; r++) {
      int t = tbase + rbase + r;
      dst[((long)b * HW + t) * CH + oc] = (__bf16)(acc[n][r] + bias);
    }
  }
}

// ====== Kernel 2b: v GEMM, transposed output. Orientation: M=o, N=t. =======
// vT stored bf16 [b][c][t].
__global__ __launch_bounds__(256) void v_gemm(const float* __restrict__ xn,
                                              const float* __restrict__ qkv_w,
                                              const float* __restrict__ qkv_b,
                                              __bf16* __restrict__ vT) {
  int wid = (int)((blockIdx.x * blockDim.x + threadIdx.x) >> 5);
  int b = wid >> 10;            // 1024 waves per batch
  int rem = wid & 1023;
  int obase = (rem & 15) * 16;  // 16 o-tiles of 16 (v channels 0..255)
  int tbase = (rem >> 4) * 64;  // 64 t-tiles of 64
  int lane = lane_id();
  f32x8 acc[4];
#pragma unroll
  for (int n = 0; n < 4; n++) zero8(acc[n]);
  const float* xnb = xn + (long)b * CH * HW;
#pragma unroll
  for (int ks = 0; ks < 8; ks++) {
    int c0 = ks * 32;
    bf16x16 af = fragA_f32(qkv_w + (long)(2 * CH + obase + (lane & 15)) * CH + c0);
#pragma unroll
    for (int n = 0; n < 4; n++) {
      bf16x16 bf = fragB_f32_gather(xnb + (long)c0 * HW + tbase + n * 16, HW);
      acc[n] = wmma_bf16(af, bf, acc[n]);
    }
  }
  int rbase = (lane < 16) ? 0 : 8;
#pragma unroll
  for (int r = 0; r < 8; r++) {
    int oo = obase + rbase + r;
    float bias = qkv_b[2 * CH + oo];
#pragma unroll
    for (int n = 0; n < 4; n++) {
      int t = tbase + n * 16 + (lane & 15);
      vT[((long)b * CH + oo) * HW + t] = (__bf16)(acc[n][r] + bias);
    }
  }
}

// ====== Kernel 3: flash attention. One wave owns a 16-row query tile. ======
__global__ __launch_bounds__(256) void flash_attn(const __bf16* __restrict__ q,
                                                  const __bf16* __restrict__ k,
                                                  const __bf16* __restrict__ vT,
                                                  __bf16* __restrict__ attno) {
  int wave = (int)(threadIdx.x >> 5);
  int wid = (int)blockIdx.x * 8 + wave;
  int b = wid >> 8;             // 256 q-tiles per batch
  int qbase = (wid & 255) * 16;
  int lane = lane_id();
  int rbase = (lane < 16) ? 0 : 8;

  __shared__ __bf16 pstage[8][16 * 40];  // per-wave 16 rows x 32 cols, 80B rows
  __bf16* pl = pstage[wave];

  // Q fragments (A-side): row = query token
  bf16x16 qf[8];
  const __bf16* qrow = q + ((long)b * HW + qbase + (lane & 15)) * CH;
#pragma unroll
  for (int kc = 0; kc < 8; kc++) qf[kc] = fragA_b16(qrow + kc * 32);

  f32x8 acc[16];
#pragma unroll
  for (int ct = 0; ct < 16; ct++) zero8(acc[ct]);
  float mrow[8], lrow[8];
#pragma unroll
  for (int r = 0; r < 8; r++) { mrow[r] = -1e30f; lrow[r] = 0.0f; }

  const __bf16* kb = k + (long)b * HW * CH;
  const __bf16* vb = vT + (long)b * CH * HW;

  for (int m0 = 0; m0 < HW; m0 += 32) {
    // scores for 32 keys: two 16-wide C tiles
    f32x8 s0, s1;
    zero8(s0); zero8(s1);
    const __bf16* krow0 = kb + (long)(m0 + (lane & 15)) * CH;
    const __bf16* krow1 = kb + (long)(m0 + 16 + (lane & 15)) * CH;
#pragma unroll
    for (int kc = 0; kc < 8; kc++) s0 = wmma_bf16(qf[kc], fragB_b16(krow0 + kc * 32), s0);
#pragma unroll
    for (int kc = 0; kc < 8; kc++) s1 = wmma_bf16(qf[kc], fragB_b16(krow1 + kc * 32), s1);

    // online softmax update (row stats live per half-wave)
    float p0[8], p1[8];
#pragma unroll
    for (int r = 0; r < 8; r++) {
      float a = s0[r] * ATT_SCALE;
      float c2 = s1[r] * ATT_SCALE;
      float mx = fmaxf(a, c2);
#pragma unroll
      for (int msk = 1; msk < 16; msk <<= 1) mx = fmaxf(mx, __shfl_xor(mx, msk, 32));
      float mnew = fmaxf(mrow[r], mx);
      float sc = __expf(mrow[r] - mnew);
      float e0 = __expf(a - mnew);
      float e1 = __expf(c2 - mnew);
      float sum = e0 + e1;
#pragma unroll
      for (int msk = 1; msk < 16; msk <<= 1) sum += __shfl_xor(sum, msk, 32);
      lrow[r] = lrow[r] * sc + sum;
      mrow[r] = mnew;
      p0[r] = e0; p1[r] = e1;
#pragma unroll
      for (int ct = 0; ct < 16; ct++) acc[ct][r] *= sc;
    }

    // C-layout -> A-layout for P via per-wave LDS round trip (DS is in-order per wave)
#pragma unroll
    for (int r = 0; r < 8; r++) {
      int row = rbase + r;
      pl[row * 40 + (lane & 15)] = (__bf16)p0[r];
      pl[row * 40 + 16 + (lane & 15)] = (__bf16)p1[r];
    }
    bf16x16 pf;
    {
      const __bf16* pp = pl + (lane & 15) * 40;
      int coff = (lane < 16) ? 0 : 8;
      bf16x8 h0 = *(const bf16x8*)(pp + coff);
      bf16x8 h1 = *(const bf16x8*)(pp + coff + 16);
#pragma unroll
      for (int i = 0; i < 8; i++) { pf[i] = h0[i]; pf[i + 8] = h1[i]; }
    }

    // O += P x V  (V fragments from vT rows: contiguous in m)
#pragma unroll
    for (int ct = 0; ct < 16; ct++) {
      const __bf16* vp = vb + (long)(ct * 16 + (lane & 15)) * HW + m0;
      acc[ct] = wmma_bf16(pf, fragB_b16(vp), acc[ct]);
    }
  }

  // finalize and store bf16 [b][t][c]
  float inv[8];
#pragma unroll
  for (int r = 0; r < 8; r++) inv[r] = 1.0f / lrow[r];
  __bf16* ob = attno + ((long)b * HW + qbase) * CH;
#pragma unroll
  for (int ct = 0; ct < 16; ct++)
#pragma unroll
    for (int r = 0; r < 8; r++)
      ob[(long)(rbase + r) * CH + ct * 16 + (lane & 15)] = (__bf16)(acc[ct][r] * inv[r]);
}

// ====== Kernel 4: out projection + bias + residual.  M=o, N=t. =============
__global__ __launch_bounds__(256) void proj_kernel(const __bf16* __restrict__ attno,
                                                   const float* __restrict__ out_w,
                                                   const float* __restrict__ out_b,
                                                   const float* __restrict__ xn,
                                                   float* __restrict__ out) {
  int wid = (int)((blockIdx.x * blockDim.x + threadIdx.x) >> 5);
  int b = wid >> 10;            // 1024 waves per batch
  int rem = wid & 1023;
  int obase = (rem & 15) * 16;
  int tbase = (rem >> 4) * 64;
  int lane = lane_id();
  f32x8 acc[4];
#pragma unroll
  for (int n = 0; n < 4; n++) zero8(acc[n]);
  const __bf16* ab = attno + (long)b * HW * CH;
#pragma unroll
  for (int ks = 0; ks < 8; ks++) {
    int c0 = ks * 32;
    bf16x16 af = fragA_f32(out_w + (long)(obase + (lane & 15)) * CH + c0);
#pragma unroll
    for (int n = 0; n < 4; n++) {
      bf16x16 bf = fragB_b16(ab + (long)(tbase + n * 16 + (lane & 15)) * CH + c0);
      acc[n] = wmma_bf16(af, bf, acc[n]);
    }
  }
  int rbase = (lane < 16) ? 0 : 8;
#pragma unroll
  for (int r = 0; r < 8; r++) {
    int o = obase + rbase + r;
    float bias = out_b[o];
    const float* xrow = xn + ((long)b * CH + o) * HW;
    float* orow = out + ((long)b * CH + o) * HW;
#pragma unroll
    for (int n = 0; n < 4; n++) {
      int t = tbase + n * 16 + (lane & 15);
      orow[t] = acc[n][r] + bias + xrow[t];  // 64B-coalesced per half-wave
    }
  }
}

// ===========================================================================
extern "C" void kernel_launch(void* const* d_in, const int* in_sizes, int n_in,
                              void* d_out, int out_size, void* d_ws, size_t ws_size,
                              hipStream_t stream) {
  (void)in_sizes; (void)n_in; (void)out_size; (void)ws_size;
  const float* x     = (const float*)d_in[0];
  const float* gn_w  = (const float*)d_in[1];
  const float* gn_b  = (const float*)d_in[2];
  const float* qkv_w = (const float*)d_in[3];
  const float* qkv_b = (const float*)d_in[4];
  const float* out_w = (const float*)d_in[5];
  const float* out_b = (const float*)d_in[6];
  float* out = (float*)d_out;

  char* ws = (char*)d_ws;
  float*  xn    = (float*) (ws);                      // 33,554,432 B
  __bf16* q     = (__bf16*)(ws + 33554432);           // 16,777,216 B
  __bf16* kk    = (__bf16*)(ws + 50331648);           // 16,777,216 B
  __bf16* vT    = (__bf16*)(ws + 67108864);           // 16,777,216 B
  __bf16* attno = (__bf16*)(ws + 83886080);           // 16,777,216 B

  gn_kernel  <<<BATCH * 8, 256, 0, stream>>>(x, gn_w, gn_b, xn);
  qk_gemm    <<<2048,      256, 0, stream>>>(xn, qkv_w, qkv_b, q, kk);
  v_gemm     <<<1024,      256, 0, stream>>>(xn, qkv_w, qkv_b, vT);
  flash_attn <<<256,       256, 0, stream>>>(q, kk, vT, attno);
  proj_kernel<<<1024,      256, 0, stream>>>(attno, out_w, out_b, xn, out);
}
